// BottomLevelDecoderRNN_90589450207554
// MI455X (gfx1250) — compile-verified, matrix operand order checked by reading, not run on hardware
//
#include <hip/hip_runtime.h>
#include <hip/hip_bf16.h>
#include <math.h>

// Problem constants (match reference)
#define Bsz  128
#define Lsz  256
#define Dsz  256
#define Hsz  256
#define Vsz  130
#define Vpad 144
#define NB   17

// Launch shape for persistent kernel
#define NWG  48
#define TPB  256
#define NWAVES ((NWG * TPB) / 32)   // 384 waves

typedef __attribute__((ext_vector_type(16))) _Float16 v16h;
typedef __attribute__((ext_vector_type(8)))  _Float16 v8h;
typedef __attribute__((ext_vector_type(8)))  float    v8f;

// ---------------------------------------------------------------------------
// WMMA helpers
// ---------------------------------------------------------------------------
__device__ __forceinline__ v8f wmma_f16(v16h a, v16h b, v8f c) {
  return __builtin_amdgcn_wmma_f32_16x16x32_f16(
      false, a, false, b, (short)0, c, /*reuse_a=*/false, /*reuse_b=*/false);
}
// Same, but hint the matrix unit to reuse the A operand (previous WMMA is the
// identical instruction with the same A fragment -> OPSEL RA legal).
__device__ __forceinline__ v8f wmma_f16_ra(v16h a, v16h b, v8f c) {
  return __builtin_amdgcn_wmma_f32_16x16x32_f16(
      false, a, false, b, (short)0, c, /*reuse_a=*/true, /*reuse_b=*/false);
}

// Load a 16x32 f16 fragment from a row-major [rows x ld] f16 matrix.
// Works for A (rows = M) and for B when weights are stored [N][K] row-major
// (torch layout), since B[k][n] = W[n][k].
__device__ __forceinline__ v16h load_frag(const _Float16* base, int ld,
                                          int row0, int k0, int lane) {
  int r  = row0 + (lane & 15);
  int kh = (lane >> 4) << 3;
  const _Float16* p = base + (size_t)r * ld + (k0 + kh);
  v8h lo = *reinterpret_cast<const v8h*>(p);        // K = k0+kh .. +7
  v8h hi = *reinterpret_cast<const v8h*>(p + 16);   // K = k0+16+kh .. +7
  v16h o;
#pragma unroll
  for (int j = 0; j < 8; ++j) { o[j] = lo[j]; o[8 + j] = hi[j]; }
  return o;
}

// ---------------------------------------------------------------------------
// Fast transcendentals: gfx1250 has a hardware v_tanh_f32 TRANS op.
// ---------------------------------------------------------------------------
__device__ __forceinline__ float fast_tanh(float x) {
#if __has_builtin(__builtin_amdgcn_tanhf)
  return __builtin_amdgcn_tanhf(x);
#else
  float e = __expf(2.0f * x);
  return (e - 1.0f) / (e + 1.0f);
#endif
}
__device__ __forceinline__ float sigm(float x) {
#if __has_builtin(__builtin_amdgcn_tanhf)
  return fmaf(0.5f, __builtin_amdgcn_tanhf(0.5f * x), 0.5f);
#else
  return 1.0f / (1.0f + __expf(-x));
#endif
}

// A K-segment of a fused (concat) GEMM held in GLOBAL memory.
struct Seg {
  const _Float16* A; int lda; int kOff; int K;
  const _Float16* W; int ldw;
};

// One wave computes a 16x16 (m0,h0) tile of ALL FOUR gates, then performs the
// LSTM pointwise update in registers. The c_t segment (if ctA != nullptr) is
// read from LDS on a dedicated path (kept out of Seg so the compiler can
// lower it to ds_load_b128); it occupies W columns [Dsz, 2*Dsz).
__device__ __forceinline__ void lstm_tile(
    int m0, int h0, int lane, const Seg* segs, int nseg,
    const _Float16* ctA, const _Float16* ctW,     // LDS c_t segment (optional)
    const float* bias,                 // [4H] combined (b_ih + b_hh)
    float* cState,                     // [B][H] f32, in-place
    _Float16* hOut,                    // [B][H] f16
    const _Float16* hAdd, _Float16* hSum) {  // optional: hSum = hOut + hAdd
  int col = h0 + (lane & 15);
  v8f acc[4];
#pragma unroll
  for (int q = 0; q < 4; ++q) {
    float bv = bias[q * Hsz + col];
#pragma unroll
    for (int r = 0; r < 8; ++r) acc[q][r] = bv;
  }
  if (ctA) {   // c_t from LDS, weights columns [Dsz, 2*Dsz) of ctW
    for (int k = 0; k < Dsz; k += 32) {
      v16h a = load_frag(ctA, Dsz, m0, k, lane);
#pragma unroll
      for (int q = 0; q < 4; ++q) {
        v16h b = load_frag(ctW, 2 * Dsz, q * Hsz + h0, Dsz + k, lane);
        acc[q] = (q == 0) ? wmma_f16(a, b, acc[q]) : wmma_f16_ra(a, b, acc[q]);
      }
    }
  }
  for (int s = 0; s < nseg; ++s) {
    Seg sg = segs[s];
    for (int k = 0; k < sg.K; k += 32) {
      v16h a = load_frag(sg.A, sg.lda, m0, k, lane);
#pragma unroll
      for (int q = 0; q < 4; ++q) {
        v16h b = load_frag(sg.W, sg.ldw, q * Hsz + h0, sg.kOff + k, lane);
        acc[q] = (q == 0) ? wmma_f16(a, b, acc[q]) : wmma_f16_ra(a, b, acc[q]);
      }
    }
  }
  int rbase = m0 + ((lane >> 4) << 3);
#pragma unroll
  for (int r = 0; r < 8; ++r) {
    int row = rbase + r;
    size_t idx = (size_t)row * Hsz + col;
    float iv = sigm(acc[0][r]);
    float fv = sigm(acc[1][r]);
    float gv = fast_tanh(acc[2][r]);
    float ov = sigm(acc[3][r]);
    float c2 = fv * cState[idx] + iv * gv;
    float h  = ov * fast_tanh(c2);
    cState[idx] = c2;
    hOut[idx] = (_Float16)h;
    if (hSum) hSum[idx] = (_Float16)(h + (float)hAdd[idx]);
  }
}

// tanh(A @ W^T + b) tile -> f16 (used for h_init, N restricted to first H)
__device__ __forceinline__ void tanh_gemm_tile(
    int m0, int n0, int lane, const _Float16* A, int lda,
    const _Float16* W, int ldw, int K, const float* bias,
    _Float16* out, int ldo) {
  int col = n0 + (lane & 15);
  v8f acc;
#pragma unroll
  for (int r = 0; r < 8; ++r) acc[r] = bias[col];
  for (int k = 0; k < K; k += 32) {
    v16h a = load_frag(A, lda, m0, k, lane);
    v16h b = load_frag(W, ldw, n0, k, lane);
    acc = wmma_f16(a, b, acc);
  }
  int rbase = m0 + ((lane >> 4) << 3);
#pragma unroll
  for (int r = 0; r < 8; ++r)
    out[(size_t)(rbase + r) * ldo + col] = (_Float16)fast_tanh(acc[r]);
}

// Logits tile: (h1+h2) @ Wout^T + bout -> f32 at out[i][row][t][col]
__device__ __forceinline__ void out_gemm_tile(
    int m0, int n0, int lane, const _Float16* A,
    const _Float16* W, const float* bias, float* dst, int i, int t) {
  int col = n0 + (lane & 15);
  v8f acc;
#pragma unroll
  for (int r = 0; r < 8; ++r) acc[r] = bias[(size_t)i * Vpad + col];
  for (int k = 0; k < Hsz; k += 32) {
    v16h a = load_frag(A, Hsz, m0, k, lane);
    v16h b = load_frag(W, Hsz, n0, k, lane);
    acc = wmma_f16(a, b, acc);
  }
  int rbase = m0 + ((lane >> 4) << 3);
  if (col < Vsz) {
#pragma unroll
    for (int r = 0; r < 8; ++r) {
      int row = rbase + r;
      dst[(((size_t)i * Bsz + row) * Lsz + t) * Vsz + col] = acc[r];
    }
  }
}

// ---------------------------------------------------------------------------
// Device-wide sense-reversal barrier (counter + generation in global memory)
// ---------------------------------------------------------------------------
__device__ __forceinline__ void grid_sync(unsigned* bar) {
  __syncthreads();
  if (threadIdx.x == 0) {
    __threadfence();
    unsigned gen = __hip_atomic_load(bar + 1, __ATOMIC_ACQUIRE,
                                     __HIP_MEMORY_SCOPE_AGENT);
    unsigned prev = __hip_atomic_fetch_add(bar, 1u, __ATOMIC_ACQ_REL,
                                           __HIP_MEMORY_SCOPE_AGENT);
    if (prev == (unsigned)(NWG - 1)) {
      __hip_atomic_store(bar, 0u, __ATOMIC_RELAXED, __HIP_MEMORY_SCOPE_AGENT);
      __hip_atomic_fetch_add(bar + 1, 1u, __ATOMIC_ACQ_REL,
                             __HIP_MEMORY_SCOPE_AGENT);
    } else {
      while (__hip_atomic_load(bar + 1, __ATOMIC_ACQUIRE,
                               __HIP_MEMORY_SCOPE_AGENT) == gen) {
        __builtin_amdgcn_s_sleep(2);
      }
    }
  }
  __syncthreads();
}

// ---------------------------------------------------------------------------
// Weight/activation conversion (fp32 -> f16, combine biases, pad Wout)
// ---------------------------------------------------------------------------
struct ConvP {
  const float *c, *Whid, *W1ih, *W1hh, *b1ih, *b1hh;
  const float *Wcih, *Wchh, *bcih, *bchh, *emb, *Wout, *bout;
  _Float16 *c16, *Whid16, *W1ih16, *W1hh16, *Wcih16, *Wchh16, *Wout16, *emb16;
  float *b1c, *bcc, *boutp;
  unsigned* bar;
};

__global__ void convert_kernel(ConvP q) {
  long tid = (long)blockIdx.x * blockDim.x + threadIdx.x;
  long stride = (long)gridDim.x * blockDim.x;
  // context: [B][NB][D] -> [NB][B][D] f16 (so each bar slice is a GEMM A)
  for (long idx = tid; idx < (long)Bsz * NB * Dsz; idx += stride) {
    long b = idx / (NB * Dsz);
    long rem = idx % (NB * Dsz);
    long bar = rem / Dsz, d = rem % Dsz;
    q.c16[(bar * Bsz + b) * Dsz + d] = (_Float16)q.c[idx];
  }
  for (long idx = tid; idx < 512L * 256; idx += stride)
    q.Whid16[idx] = (_Float16)q.Whid[idx];
  for (long idx = tid; idx < 3L * 1024 * 512; idx += stride)
    q.W1ih16[idx] = (_Float16)q.W1ih[idx];
  for (long idx = tid; idx < 3L * 1024 * 256; idx += stride)
    q.W1hh16[idx] = (_Float16)q.W1hh[idx];
  for (long idx = tid; idx < 1024L * 768; idx += stride)
    q.Wcih16[idx] = (_Float16)q.Wcih[idx];
  for (long idx = tid; idx < 1024L * 256; idx += stride)
    q.Wchh16[idx] = (_Float16)q.Wchh[idx];
  for (long idx = tid; idx < 3L * 1024; idx += stride)
    q.b1c[idx] = q.b1ih[idx] + q.b1hh[idx];
  for (long idx = tid; idx < 1024L; idx += stride)
    q.bcc[idx] = q.bcih[idx] + q.bchh[idx];
  for (long idx = tid; idx < 3L * Vpad * Dsz; idx += stride) {
    long i = idx / ((long)Vpad * Dsz);
    long n = (idx / Dsz) % Vpad;
    long k = idx % Dsz;
    q.Wout16[idx] = (n < Vsz) ? (_Float16)q.Wout[(i * Vsz + n) * Dsz + k]
                              : (_Float16)0.0f;
  }
  for (long idx = tid; idx < 3L * Vpad; idx += stride) {
    long i = idx / Vpad, n = idx % Vpad;
    q.boutp[idx] = (n < Vsz) ? q.bout[i * Vsz + n] : 0.0f;
  }
  for (long idx = tid; idx < 3L * Vsz * Dsz; idx += stride)
    q.emb16[idx] = (_Float16)q.emb[idx];
  if (tid == 0) { q.bar[0] = 0u; q.bar[1] = 0u; }
}

// ---------------------------------------------------------------------------
// Persistent decoder kernel: whole L=256 time loop, grid barriers between
// dependent GEMM stages. 384 waves == 3*8*16 gate tiles of the lstm1 stage.
// c_t slice is staged in LDS once per bar (reused ~144x from LDS).
// ---------------------------------------------------------------------------
struct DecP {
  const int* target;               // [3][B][L] int32
  const _Float16 *c16;             // [NB][B][D]
  const _Float16 *Whid, *W1ih, *W1hh, *Wcih, *Wchh, *Wout, *emb;
  const float *bhid, *b1, *bc, *bout;
  float *c1, *c2s, *cc;            // f32 cell states (in-place)
  _Float16 *h1, *h2, *hc;          // double buffered: [2][...]
  _Float16 *notes, *hsum, *hinit;
  unsigned* bar;
  float* out;                      // [3][B][L][V]
};

__global__ __launch_bounds__(TPB, 1) void decoder_kernel(DecP p) {
  __shared__ _Float16 ctLDS[Bsz * Dsz];   // 64 KB of the 320 KB WGP LDS

  const int gtid = blockIdx.x * TPB + threadIdx.x;
  const int nthreads = NWG * TPB;
  const int gwave = gtid >> 5;
  const int lane = threadIdx.x & 31;

  // Warm L2 with the hot weights (global_prefetch_b8).
  {
    const char* w = (const char*)p.W1ih;
    size_t bytes = 3ull * 1024 * 512 * sizeof(_Float16);
    for (size_t o = (size_t)gtid * 128; o < bytes; o += (size_t)nthreads * 128)
      __builtin_prefetch(w + o, 0, 1);
  }

  // notes <- emb[:,0,:]
  for (int idx = gtid; idx < 3 * Bsz * Dsz; idx += nthreads) {
    int i = idx / (Bsz * Dsz);
    int d = idx % Dsz;
    p.notes[idx] = p.emb[((size_t)i * Vsz + 0) * Dsz + d];
  }
  grid_sync(p.bar);

  int h1cur[3] = {0, 0, 0};
  int h2cur[3] = {0, 0, 0};
  int hccur = 0;
  const size_t S = (size_t)Bsz * Hsz;           // one [B][H] slab
  auto h1p = [&](int i, int c) { return p.h1 + ((size_t)c * 3 + i) * S; };
  auto h2p = [&](int i, int c) { return p.h2 + ((size_t)c * 3 + i) * S; };
  auto hcp = [&](int c)        { return p.hc + (size_t)c * S; };

  for (int t = 0; t < Lsz; ++t) {
    int barid = t >> 4;
    const _Float16* ct = p.c16 + (size_t)(barid + 1) * Bsz * Dsz;

    if ((t & 15) == 0) {
      // Stage c_t (next-bar context) into LDS: global_load_b128 + ds_store.
      for (int idx = threadIdx.x * 8; idx < Bsz * Dsz; idx += TPB * 8)
        *reinterpret_cast<v8h*>(&ctLDS[idx]) =
            *reinterpret_cast<const v8h*>(&ct[idx]);
      __syncthreads();

      // h_init = tanh(c_bar @ W_hid^T + b_hid)[:, :H]
      const _Float16* cb = p.c16 + (size_t)barid * Bsz * Dsz;
      for (int task = gwave; task < 8 * 16; task += NWAVES) {
        int m0 = (task >> 4) << 4, n0 = (task & 15) << 4;
        tanh_gemm_tile(m0, n0, lane, cb, Dsz, p.Whid, Dsz, Dsz,
                       p.bhid, p.hinit, Hsz);
      }
      grid_sync(p.bar);
      // reset: h <- h_init (into *current* buffers), c <- 0
      for (int idx = gtid; idx < Bsz * Hsz; idx += nthreads) {
        _Float16 h = p.hinit[idx];
        hcp(hccur)[idx] = h; p.cc[idx] = 0.0f;
        for (int i = 0; i < 3; ++i) {
          h1p(i, h1cur[i])[idx] = h; p.c1[(size_t)i * S + idx] = 0.0f;
          h2p(i, h2cur[i])[idx] = h; p.c2s[(size_t)i * S + idx] = 0.0f;
        }
      }
      grid_sync(p.bar);
    }

    // ---- lstm1 for all 3 streams: x = [notes_i | c_t], h = h1_i ----
    for (int task = gwave; task < 3 * 8 * 16; task += NWAVES) {
      int i = task / 128, r = task % 128;
      int m0 = (r >> 4) << 4, h0 = (r & 15) << 4;
      const _Float16* Wih = p.W1ih + (size_t)i * 4 * Hsz * (2 * Dsz);
      const _Float16* Whh = p.W1hh + (size_t)i * 4 * Hsz * Hsz;
      Seg segs[2] = {
          { p.notes + (size_t)i * S, Dsz, 0, Dsz, Wih, 2 * Dsz },
          { h1p(i, h1cur[i]),        Hsz, 0, Hsz, Whh, Hsz     },
      };
      lstm_tile(m0, h0, lane, segs, 2, ctLDS, Wih,
                p.b1 + (size_t)i * 4 * Hsz,
                p.c1 + (size_t)i * S, h1p(i, 1 - h1cur[i]),
                nullptr, nullptr);
    }
    grid_sync(p.bar);
    h1cur[0] ^= 1; h1cur[1] ^= 1; h1cur[2] ^= 1;

    for (int i = 0; i < 3; ++i) {
      // ---- context LSTM: x = [h1_0|h1_1|h1_2] (no c_t segment) ----
      for (int task = gwave; task < 8 * 16; task += NWAVES) {
        int m0 = (task >> 4) << 4, h0 = (task & 15) << 4;
        Seg segs[4] = {
            { h1p(0, h1cur[0]), Hsz, 0,       Hsz, p.Wcih, 3 * Hsz },
            { h1p(1, h1cur[1]), Hsz, Hsz,     Hsz, p.Wcih, 3 * Hsz },
            { h1p(2, h1cur[2]), Hsz, 2 * Hsz, Hsz, p.Wcih, 3 * Hsz },
            { hcp(hccur),       Hsz, 0,       Hsz, p.Wchh, Hsz     },
        };
        lstm_tile(m0, h0, lane, segs, 4, nullptr, nullptr,
                  p.bc, p.cc, hcp(1 - hccur), nullptr, nullptr);
      }
      grid_sync(p.bar);
      hccur ^= 1;

      // ---- h2_i LSTM: x = [hc | c_t]; also hsum = h1_i + h2_i ----
      {
        const _Float16* Wih = p.W1ih + (size_t)i * 4 * Hsz * (2 * Dsz);
        const _Float16* Whh = p.W1hh + (size_t)i * 4 * Hsz * Hsz;
        for (int task = gwave; task < 8 * 16; task += NWAVES) {
          int m0 = (task >> 4) << 4, h0 = (task & 15) << 4;
          Seg segs[2] = {
              { hcp(hccur),       Hsz, 0, Hsz, Wih, 2 * Dsz },
              { h2p(i, h2cur[i]), Hsz, 0, Hsz, Whh, Hsz     },
          };
          lstm_tile(m0, h0, lane, segs, 2, ctLDS, Wih,
                    p.b1 + (size_t)i * 4 * Hsz,
                    p.c2s + (size_t)i * S, h2p(i, 1 - h2cur[i]),
                    h1p(i, h1cur[i]), p.hsum);
        }
        grid_sync(p.bar);
        h2cur[i] ^= 1;
      }

      // ---- logits GEMM on hsum + teacher-forcing embedding gather ----
      for (int task = gwave; task < 8 * (Vpad / 16); task += NWAVES) {
        int m0 = (task / (Vpad / 16)) << 4;
        int n0 = (task % (Vpad / 16)) << 4;
        out_gemm_tile(m0, n0, lane, p.hsum,
                      p.Wout + (size_t)i * Vpad * Dsz, p.bout, p.out, i, t);
      }
      for (int idx = gtid; idx < Bsz * Dsz; idx += nthreads) {
        int b = idx / Dsz, d = idx % Dsz;
        int tok = p.target[((size_t)i * Bsz + b) * Lsz + t];
        p.notes[(size_t)i * S + idx] =
            p.emb[((size_t)i * Vsz + tok) * Dsz + d];
      }
      grid_sync(p.bar);

      // ---- h1_i re-update: x = [notes_i(new) | c_t] ----
      {
        const _Float16* Wih = p.W1ih + (size_t)i * 4 * Hsz * (2 * Dsz);
        const _Float16* Whh = p.W1hh + (size_t)i * 4 * Hsz * Hsz;
        for (int task = gwave; task < 8 * 16; task += NWAVES) {
          int m0 = (task >> 4) << 4, h0 = (task & 15) << 4;
          Seg segs[2] = {
              { p.notes + (size_t)i * S, Dsz, 0, Dsz, Wih, 2 * Dsz },
              { h1p(i, h1cur[i]),        Hsz, 0, Hsz, Whh, Hsz     },
          };
          lstm_tile(m0, h0, lane, segs, 2, ctLDS, Wih,
                    p.b1 + (size_t)i * 4 * Hsz,
                    p.c1 + (size_t)i * S, h1p(i, 1 - h1cur[i]),
                    nullptr, nullptr);
        }
        grid_sync(p.bar);
        h1cur[i] ^= 1;
      }
    }
  }
}

// ---------------------------------------------------------------------------
// Host launcher
// ---------------------------------------------------------------------------
extern "C" void kernel_launch(void* const* d_in, const int* in_sizes, int n_in,
                              void* d_out, int out_size, void* d_ws,
                              size_t ws_size, hipStream_t stream) {
  (void)in_sizes; (void)n_in; (void)out_size; (void)ws_size;
  const float* c      = (const float*)d_in[0];
  const int*   target = (const int*)d_in[1];
  const float* W_hid  = (const float*)d_in[3];
  const float* b_hid  = (const float*)d_in[4];
  const float* W1_ih  = (const float*)d_in[5];
  const float* W1_hh  = (const float*)d_in[6];
  const float* b1_ih  = (const float*)d_in[7];
  const float* b1_hh  = (const float*)d_in[8];
  const float* Wc_ih  = (const float*)d_in[9];
  const float* Wc_hh  = (const float*)d_in[10];
  const float* bc_ih  = (const float*)d_in[11];
  const float* bc_hh  = (const float*)d_in[12];
  const float* emb    = (const float*)d_in[13];
  const float* Wout   = (const float*)d_in[14];
  const float* bout   = (const float*)d_in[15];

  char* w = (char*)d_ws;
  size_t off = 0;
  auto take = [&](size_t bytes) -> char* {
    char* p = w + off;
    off += (bytes + 255) & ~(size_t)255;
    return p;
  };

  _Float16* c16    = (_Float16*)take((size_t)NB * Bsz * Dsz * 2);
  _Float16* Whid16 = (_Float16*)take(512ull * 256 * 2);
  _Float16* W1ih16 = (_Float16*)take(3ull * 1024 * 512 * 2);
  _Float16* W1hh16 = (_Float16*)take(3ull * 1024 * 256 * 2);
  _Float16* Wcih16 = (_Float16*)take(1024ull * 768 * 2);
  _Float16* Wchh16 = (_Float16*)take(1024ull * 256 * 2);
  _Float16* Wout16 = (_Float16*)take(3ull * Vpad * Dsz * 2);
  _Float16* emb16  = (_Float16*)take(3ull * Vsz * Dsz * 2);
  float*    b1c    = (float*)take(3ull * 1024 * 4);
  float*    bcc    = (float*)take(1024ull * 4);
  float*    boutp  = (float*)take(3ull * Vpad * 4);
  float*    c1     = (float*)take(3ull * Bsz * Hsz * 4);
  float*    c2s    = (float*)take(3ull * Bsz * Hsz * 4);
  float*    cc     = (float*)take((size_t)Bsz * Hsz * 4);
  _Float16* h1     = (_Float16*)take(2ull * 3 * Bsz * Hsz * 2);
  _Float16* h2     = (_Float16*)take(2ull * 3 * Bsz * Hsz * 2);
  _Float16* hc     = (_Float16*)take(2ull * Bsz * Hsz * 2);
  _Float16* notes  = (_Float16*)take(3ull * Bsz * Dsz * 2);
  _Float16* hsum   = (_Float16*)take((size_t)Bsz * Hsz * 2);
  _Float16* hinit  = (_Float16*)take((size_t)Bsz * Hsz * 2);
  unsigned* bar    = (unsigned*)take(2 * sizeof(unsigned));

  ConvP cp;
  cp.c = c; cp.Whid = W_hid; cp.W1ih = W1_ih; cp.W1hh = W1_hh;
  cp.b1ih = b1_ih; cp.b1hh = b1_hh; cp.Wcih = Wc_ih; cp.Wchh = Wc_hh;
  cp.bcih = bc_ih; cp.bchh = bc_hh; cp.emb = emb; cp.Wout = Wout;
  cp.bout = bout;
  cp.c16 = c16; cp.Whid16 = Whid16; cp.W1ih16 = W1ih16; cp.W1hh16 = W1hh16;
  cp.Wcih16 = Wcih16; cp.Wchh16 = Wchh16; cp.Wout16 = Wout16;
  cp.emb16 = emb16; cp.b1c = b1c; cp.bcc = bcc; cp.boutp = boutp;
  cp.bar = bar;
  convert_kernel<<<512, 256, 0, stream>>>(cp);

  DecP pp;
  pp.target = target; pp.c16 = c16;
  pp.Whid = Whid16; pp.W1ih = W1ih16; pp.W1hh = W1hh16;
  pp.Wcih = Wcih16; pp.Wchh = Wchh16; pp.Wout = Wout16; pp.emb = emb16;
  pp.bhid = b_hid; pp.b1 = b1c; pp.bc = bcc; pp.bout = boutp;
  pp.c1 = c1; pp.c2s = c2s; pp.cc = cc;
  pp.h1 = h1; pp.h2 = h2; pp.hc = hc;
  pp.notes = notes; pp.hsum = hsum; pp.hinit = hinit;
  pp.bar = bar; pp.out = (float*)d_out;
  decoder_kernel<<<NWG, TPB, 0, stream>>>(pp);
}
